// Attention_49907519980190
// MI455X (gfx1250) — matrix operand order repaired and checked by reading, not compiled, not run on hardware
//
#include <hip/hip_runtime.h>

// ---------------------------------------------------------------------------
// CDNA5 (gfx1250) bf16 WMMA attention: 3-kernel pipeline, all GEMMs on
// v_wmma_f32_16x16x32_bf16 (wave32, 16x16 f32 accumulators in 8 VGPRs).
// Attention K/V staging uses GLOBAL_LOAD_ASYNC_TO_LDS_B128 (ASYNCcnt) when
// the toolchain exposes the builtins, else falls back to b128 loads+stores.
// ---------------------------------------------------------------------------

typedef __bf16 bf16_t;
typedef __attribute__((ext_vector_type(16))) __bf16 v16bf;
typedef __attribute__((ext_vector_type(8)))  float  v8f;
typedef int v4i __attribute__((vector_size(16)));   // matches builtin param type

#if defined(__gfx1250__) && \
    __has_builtin(__builtin_amdgcn_global_load_async_to_lds_b128) && \
    __has_builtin(__builtin_amdgcn_s_wait_asynccnt)
#define USE_ASYNC_LDS 1
// global (addrspace 1) / LDS (addrspace 3) typed-pointer casts
#define ASYNC_GPTR(p) ((__attribute__((address_space(1))) v4i*)(p))
#define ASYNC_LPTR(p) ((__attribute__((address_space(3))) v4i*)(p))
#else
#define USE_ASYNC_LDS 0
#endif

union Frag16 { v16bf v; uint4 u[2]; };

// A-fragment (16x32, M x K): lane m = lane&15, half h = lane>>4.
// elements 0..7  <- K = h*8 + 0..7
// elements 8..15 <- K = 16 + h*8 + 0..7
__device__ __forceinline__ v16bf load_a_frag(const bf16_t* base, int lda, int lane) {
  const int h = (lane >> 4) & 1, m = lane & 15;
  const bf16_t* p = base + m * lda + h * 8;
  Frag16 f;
  f.u[0] = *reinterpret_cast<const uint4*>(p);
  f.u[1] = *reinterpret_cast<const uint4*>(p + 16);
  return f.v;
}

// B-fragment (32x16, K x N) read from a TRANSPOSED operand Bt[n][k]:
// lane n = lane&15, half h = lane>>4; elements 0..7 <- K = h*8 + 0..7, 8..15 <- K = 16 + h*8 + 0..7.
__device__ __forceinline__ v16bf load_bt_frag(const bf16_t* base, int ldb, int lane) {
  const int h = (lane >> 4) & 1, n = lane & 15;
  const bf16_t* p = base + n * ldb + h * 8;
  Frag16 f;
  f.u[0] = *reinterpret_cast<const uint4*>(p);
  f.u[1] = *reinterpret_cast<const uint4*>(p + 16);
  return f.v;
}

__device__ __forceinline__ v8f wmma_bf16(v16bf a, v16bf b, v8f c) {
  return __builtin_amdgcn_wmma_f32_16x16x32_bf16(
      /*neg_a=*/false, a, /*neg_b=*/false, b,
      /*c_mod=*/(short)0, c, /*reuse_a=*/false, /*reuse_b=*/false);
}

// ---------------------------------------------------------------------------
// Problem constants
// ---------------------------------------------------------------------------
#define N_TOK   2048
#define DIM     1024
#define HEADS   16
#define DHEAD   64
#define BATCH   2
#define M_TOT   (BATCH * N_TOK)     // 4096
#define SCALE_Q 8.0f                // sqrt(DHEAD), reference MULTIPLIES

// GEMM tiling: 256 threads = 8 waves arranged 4(M) x 2(N).
// Workgroup tile 128x128, wave tile 32x64, K-step 32.
#define BM 128
#define BN 128
#define KS 32

// ---------------------------------------------------------------------------
// Kernel 1: fused QKV projection.
//   Q[b][h][n][d]  = (x @ w_q) * 8      (pre-scaled, bf16)
//   K[b][h][n][d]  =  x @ w_vk[:, :1024]
//   Vt[b][h][d][n] = (x @ w_vk[:, 1024:]) transposed
// Output column space c in [0,3072): [0,1024)->Q, [1024,2048)->K, rest->V.
// ---------------------------------------------------------------------------
__global__ __launch_bounds__(256) void qkv_proj_kernel(
    const float* __restrict__ x,     // 4096 x 1024
    const float* __restrict__ w_q,   // 1024 x 1024
    const float* __restrict__ w_vk,  // 1024 x 2048
    bf16_t* __restrict__ Q, bf16_t* __restrict__ K, bf16_t* __restrict__ Vt)
{
  __shared__ __align__(16) bf16_t lds_a[BM * KS];    // x tile, row-major
  __shared__ __align__(16) bf16_t lds_bt[BN * KS];   // weight tile, TRANSPOSED

  const int tid = threadIdx.x, lane = tid & 31, wave = tid >> 5;
  const int wm = wave >> 1;            // 0..3 -> M offset wm*32
  const int wn = wave & 1;             // 0..1 -> N offset wn*64
  const int m0 = blockIdx.x * BM;
  const int c0 = blockIdx.y * BN;      // global output column base

  const float* W; int ldw, wc0;
  if (c0 < 1024) { W = w_q;  ldw = 1024; wc0 = c0; }
  else           { W = w_vk; ldw = 2048; wc0 = c0 - 1024; }

  v8f acc[2][4] = {};

  for (int kb = 0; kb < DIM; kb += KS) {
    __syncthreads();
    {   // stage A: 128x32 fp32 -> bf16 (16 elems/thread)
      const int r = tid >> 1, c = (tid & 1) * 16;
      const float* src = x + (size_t)(m0 + r) * DIM + kb + c;
      __align__(16) bf16_t t[16];
      #pragma unroll
      for (int i = 0; i < 16; ++i) t[i] = (bf16_t)src[i];
      *reinterpret_cast<uint4*>(&lds_a[r * KS + c])     = *reinterpret_cast<uint4*>(&t[0]);
      *reinterpret_cast<uint4*>(&lds_a[r * KS + c + 8]) = *reinterpret_cast<uint4*>(&t[8]);
    }
    {   // stage Bt: read W[k][n] (lane-coalesced along n), write lds_bt[n][k]
      const int n = tid & 127, kh = (tid >> 7) * 16;
      __align__(16) bf16_t t[16];
      #pragma unroll
      for (int i = 0; i < 16; ++i)
        t[i] = (bf16_t)W[(size_t)(kb + kh + i) * ldw + wc0 + n];
      *reinterpret_cast<uint4*>(&lds_bt[n * KS + kh])     = *reinterpret_cast<uint4*>(&t[0]);
      *reinterpret_cast<uint4*>(&lds_bt[n * KS + kh + 8]) = *reinterpret_cast<uint4*>(&t[8]);
    }
    __syncthreads();

    v16bf af[2], bf[4];
    #pragma unroll
    for (int i = 0; i < 2; ++i)
      af[i] = load_a_frag(&lds_a[(wm * 32 + i * 16) * KS], KS, lane);
    #pragma unroll
    for (int j = 0; j < 4; ++j)
      bf[j] = load_bt_frag(&lds_bt[(wn * 64 + j * 16) * KS], KS, lane);
    #pragma unroll
    for (int i = 0; i < 2; ++i)
      #pragma unroll
      for (int j = 0; j < 4; ++j)
        acc[i][j] = wmma_bf16(af[i], bf[j], acc[i][j]);
  }

  // Epilogue. C-tile layout: lane col n = lane&15; rows m = r + 8*(lane>>4).
  const int h = (lane >> 4) & 1, nn = lane & 15;
  #pragma unroll
  for (int i = 0; i < 2; ++i) {
    const int row0 = m0 + wm * 32 + i * 16 + h * 8;   // 8 consecutive rows
    const int b = row0 >> 11, pos0 = row0 & (N_TOK - 1);
    #pragma unroll
    for (int j = 0; j < 4; ++j) {
      const int col = c0 + wn * 64 + j * 16 + nn;
      if (col < 1024) {                               // Q (pre-scaled by 8)
        const int hh = col >> 6, dd = col & 63;
        bf16_t* dst = Q + ((((size_t)b * HEADS + hh) * N_TOK + pos0) * DHEAD + dd);
        #pragma unroll
        for (int r = 0; r < 8; ++r) dst[r * DHEAD] = (bf16_t)(acc[i][j][r] * SCALE_Q);
      } else if (col < 2048) {                        // K
        const int c2 = col - 1024, hh = c2 >> 6, dd = c2 & 63;
        bf16_t* dst = K + ((((size_t)b * HEADS + hh) * N_TOK + pos0) * DHEAD + dd);
        #pragma unroll
        for (int r = 0; r < 8; ++r) dst[r * DHEAD] = (bf16_t)acc[i][j][r];
      } else {                                        // V, stored transposed
        const int c2 = col - 2048, hh = c2 >> 6, dd = c2 & 63;
        __align__(16) bf16_t t[8];
        #pragma unroll
        for (int r = 0; r < 8; ++r) t[r] = (bf16_t)acc[i][j][r];
        bf16_t* dst = Vt + ((((size_t)b * HEADS + hh) * DHEAD + dd) * N_TOK + pos0);
        *reinterpret_cast<uint4*>(dst) = *reinterpret_cast<uint4*>(&t[0]);
      }
    }
  }
}

// ---------------------------------------------------------------------------
// Kernel 2: flash attention per (b,h). 256 threads = 8 waves; each wave owns
// 16 query rows; K/Vt staged in 64-key chunks shared by the workgroup.
// Q is pre-scaled by sqrt(d), so S needs no extra scaling.
// ---------------------------------------------------------------------------
#define NCH 64    // key-chunk size

__global__ __launch_bounds__(256) void attn_kernel(
    const bf16_t* __restrict__ Q, const bf16_t* __restrict__ K,
    const bf16_t* __restrict__ Vt,
    bf16_t* __restrict__ O)              // (4096 x 1024): row b*2048+pos, col h*64+d
{
  __shared__ __align__(16) bf16_t lds_k[NCH * DHEAD];      // [j][d]   8 KB
  __shared__ __align__(16) bf16_t lds_v[DHEAD * NCH];      // [d][j]   8 KB
  __shared__ __align__(16) bf16_t lds_p[8][16 * NCH];      // per-wave 16 KB

  const int tid = threadIdx.x, lane = tid & 31, wave = tid >> 5;
  const int bh = blockIdx.y;                               // 0..31
  const size_t base = (size_t)bh * N_TOK * DHEAD;
  const int q0 = blockIdx.x * 128 + wave * 16;
  const int hh = (lane >> 4) & 1, nn = lane & 15;

  // Q fragments (A-layout) straight from global: rows contiguous, ld = 64.
  const bf16_t* qbase = Q + base + (size_t)q0 * DHEAD;
  v16bf qf0 = load_a_frag(qbase + 0,  DHEAD, lane);
  v16bf qf1 = load_a_frag(qbase + 32, DHEAD, lane);

  float mrow[8], lrow[8];
  #pragma unroll
  for (int r = 0; r < 8; ++r) { mrow[r] = -1e30f; lrow[r] = 0.0f; }
  v8f oacc[4] = {};

  for (int j0 = 0; j0 < N_TOK; j0 += NCH) {
    __syncthreads();
    {   // stage K chunk (64 rows x 128 B) and Vt chunk (64 d-rows, 128 B each):
        // 32 B per thread per buffer.
      const int rr = tid >> 2, cc = (tid & 3) * 16;
      const bf16_t* ksrc = K  + base + (size_t)(j0 + rr) * DHEAD + cc;
      const bf16_t* vsrc = Vt + base + (size_t)rr * N_TOK + j0 + cc;
      bf16_t* kdst = &lds_k[rr * DHEAD + cc];
      bf16_t* vdst = &lds_v[rr * NCH + cc];
#if USE_ASYNC_LDS
      __builtin_amdgcn_global_load_async_to_lds_b128(
          ASYNC_GPTR(const_cast<bf16_t*>(ksrc)), ASYNC_LPTR(kdst), 0, 0);
      __builtin_amdgcn_global_load_async_to_lds_b128(
          ASYNC_GPTR(const_cast<bf16_t*>(ksrc)), ASYNC_LPTR(kdst), 16, 0);
      __builtin_amdgcn_global_load_async_to_lds_b128(
          ASYNC_GPTR(const_cast<bf16_t*>(vsrc)), ASYNC_LPTR(vdst), 0, 0);
      __builtin_amdgcn_global_load_async_to_lds_b128(
          ASYNC_GPTR(const_cast<bf16_t*>(vsrc)), ASYNC_LPTR(vdst), 16, 0);
      __builtin_amdgcn_s_wait_asynccnt(0);
#else
      {
        const uint4* s0 = reinterpret_cast<const uint4*>(ksrc);
        uint4* d0 = reinterpret_cast<uint4*>(kdst);
        d0[0] = s0[0]; d0[1] = s0[1];
        const uint4* s1 = reinterpret_cast<const uint4*>(vsrc);
        uint4* d1 = reinterpret_cast<uint4*>(vdst);
        d1[0] = s1[0]; d1[1] = s1[1];
      }
#endif
    }
    if (j0 + NCH < N_TOK) {   // pull next chunk toward L2/L0 (global_prefetch_b8)
      const int rr = tid >> 2;
      __builtin_prefetch(K + base + (size_t)(j0 + NCH + rr) * DHEAD, 0, 3);
      __builtin_prefetch(Vt + base + (size_t)rr * N_TOK + j0 + NCH, 0, 3);
    }
    __syncthreads();

    // S = Q · K^T : 4 tiles of 16x16, K-dim = d = 64 (2 WMMA steps each)
    v8f s[4];
    #pragma unroll
    for (int jt = 0; jt < 4; ++jt) {
      v8f z = {};
      v16bf b0 = load_bt_frag(&lds_k[(jt * 16) * DHEAD + 0],  DHEAD, lane);
      v16bf b1 = load_bt_frag(&lds_k[(jt * 16) * DHEAD + 32], DHEAD, lane);
      z = wmma_bf16(qf0, b0, z);
      s[jt] = wmma_bf16(qf1, b1, z);
    }

    // Online softmax: row m = r + 8*hh lives in one 16-lane half at VGPR r.
    float preg[4][8];
    #pragma unroll
    for (int r = 0; r < 8; ++r) {
      float tm = fmaxf(fmaxf(s[0][r], s[1][r]), fmaxf(s[2][r], s[3][r]));
      #pragma unroll
      for (int off = 8; off >= 1; off >>= 1)
        tm = fmaxf(tm, __shfl_xor(tm, off, 32));
      const float mnew = fmaxf(mrow[r], tm);
      const float corr = __expf(mrow[r] - mnew);
      mrow[r] = mnew;
      float psum = 0.0f;
      #pragma unroll
      for (int jt = 0; jt < 4; ++jt) {
        const float p = __expf(s[jt][r] - mnew);
        preg[jt][r] = p;
        psum += p;
      }
      #pragma unroll
      for (int off = 8; off >= 1; off >>= 1)
        psum += __shfl_xor(psum, off, 32);
      lrow[r] = lrow[r] * corr + psum;
      #pragma unroll
      for (int dt = 0; dt < 4; ++dt) oacc[dt][r] *= corr;
    }

    // P (C-layout) -> per-wave LDS (row-major 16x64) -> A-fragments
    bf16_t* pw = &lds_p[wave][0];
    #pragma unroll
    for (int r = 0; r < 8; ++r) {
      const int m = r + 8 * hh;
      #pragma unroll
      for (int jt = 0; jt < 4; ++jt)
        pw[m * NCH + jt * 16 + nn] = (bf16_t)preg[jt][r];
    }
    __syncthreads();

    // O += P · V : N-dim = d (4 tiles), K-dim = j (2 WMMA steps)
    #pragma unroll
    for (int ks = 0; ks < 2; ++ks) {
      v16bf pa = load_a_frag(pw + ks * 32, NCH, lane);
      #pragma unroll
      for (int dt = 0; dt < 4; ++dt) {
        v16bf vb = load_bt_frag(&lds_v[(dt * 16) * NCH + ks * 32], NCH, lane);
        oacc[dt] = wmma_bf16(pa, vb, oacc[dt]);
      }
    }
  }

  // Normalize and store O rows as bf16 in (b*n, h*d) layout for the out-proj.
  float inv[8];
  #pragma unroll
  for (int r = 0; r < 8; ++r) inv[r] = 1.0f / lrow[r];
  const int bb = bh >> 4, head = bh & (HEADS - 1);
  const size_t row0 = (size_t)bb * N_TOK + q0 + 8 * hh;
  #pragma unroll
  for (int dt = 0; dt < 4; ++dt) {
    const int col = head * DHEAD + dt * 16 + nn;
    bf16_t* dst = O + row0 * DIM + col;
    #pragma unroll
    for (int r = 0; r < 8; ++r)
      dst[(size_t)r * DIM] = (bf16_t)(oacc[dt][r] * inv[r]);
  }
}

// ---------------------------------------------------------------------------
// Kernel 3: out = O (4096x1024 bf16) @ w_out (1024x1024 f32), fp32 result.
// A fragments straight from global (already bf16, row-major); Bt staged.
// ---------------------------------------------------------------------------
__global__ __launch_bounds__(256) void out_proj_kernel(
    const bf16_t* __restrict__ O, const float* __restrict__ w_out,
    float* __restrict__ out)
{
  __shared__ __align__(16) bf16_t lds_bt[BN * KS];

  const int tid = threadIdx.x, lane = tid & 31, wave = tid >> 5;
  const int wm = wave >> 1, wn = wave & 1;
  const int m0 = blockIdx.x * BM, c0 = blockIdx.y * BN;

  v8f acc[2][4] = {};

  for (int kb = 0; kb < DIM; kb += KS) {
    __syncthreads();
    {
      const int n = tid & 127, kh = (tid >> 7) * 16;
      __align__(16) bf16_t t[16];
      #pragma unroll
      for (int i = 0; i < 16; ++i)
        t[i] = (bf16_t)w_out[(size_t)(kb + kh + i) * DIM + c0 + n];
      *reinterpret_cast<uint4*>(&lds_bt[n * KS + kh])     = *reinterpret_cast<uint4*>(&t[0]);
      *reinterpret_cast<uint4*>(&lds_bt[n * KS + kh + 8]) = *reinterpret_cast<uint4*>(&t[8]);
    }
    __syncthreads();

    v16bf af[2], bf[4];
    #pragma unroll
    for (int i = 0; i < 2; ++i)
      af[i] = load_a_frag(O + (size_t)(m0 + wm * 32 + i * 16) * DIM + kb, DIM, lane);
    #pragma unroll
    for (int j = 0; j < 4; ++j)
      bf[j] = load_bt_frag(&lds_bt[(wn * 64 + j * 16) * KS], KS, lane);
    #pragma unroll
    for (int i = 0; i < 2; ++i)
      #pragma unroll
      for (int j = 0; j < 4; ++j)
        acc[i][j] = wmma_bf16(af[i], bf[j], acc[i][j]);
  }

  const int h = (lane >> 4) & 1, nn = lane & 15;
  #pragma unroll
  for (int i = 0; i < 2; ++i) {
    const int row0 = m0 + wm * 32 + i * 16 + h * 8;
    #pragma unroll
    for (int j = 0; j < 4; ++j) {
      const int col = c0 + wn * 64 + j * 16 + nn;
      float* dst = out + (size_t)row0 * DIM + col;
      #pragma unroll
      for (int r = 0; r < 8; ++r) dst[(size_t)r * DIM] = acc[i][j][r];
    }
  }
}

// ---------------------------------------------------------------------------
// Launch: workspace carve-up (bf16): Q(8MB) | K(8MB) | Vt(8MB) | O(8MB)
// ---------------------------------------------------------------------------
extern "C" void kernel_launch(void* const* d_in, const int* in_sizes, int n_in,
                              void* d_out, int out_size, void* d_ws, size_t ws_size,
                              hipStream_t stream) {
  const float* x     = (const float*)d_in[0];
  const float* w_q   = (const float*)d_in[1];
  const float* w_vk  = (const float*)d_in[2];
  const float* w_out = (const float*)d_in[3];
  float* out = (float*)d_out;

  const size_t elems = (size_t)BATCH * HEADS * N_TOK * DHEAD;   // 4M bf16
  char* ws = (char*)d_ws;
  bf16_t* Q  = (bf16_t*)(ws);
  bf16_t* K  = (bf16_t*)(ws + 2 * elems);
  bf16_t* Vt = (bf16_t*)(ws + 4 * elems);
  bf16_t* O  = (bf16_t*)(ws + 6 * elems);

  dim3 gA(M_TOT / BM, (DIM + 2 * DIM) / BN);   // 32 x 24
  qkv_proj_kernel<<<gA, 256, 0, stream>>>(x, w_q, w_vk, Q, K, Vt);

  dim3 gB(N_TOK / 128, BATCH * HEADS);         // 16 x 32
  attn_kernel<<<gB, 256, 0, stream>>>(Q, K, Vt, O);

  dim3 gC(M_TOT / BM, DIM / BN);               // 32 x 8
  out_proj_kernel<<<gC, 256, 0, stream>>>(O, w_out, out);
}
